// UnitAGCN_3573412790771
// MI455X (gfx1250) — matrix operand order repaired and checked by reading, not compiled, not run on hardware
//
#include <hip/hip_runtime.h>

// ---------------- problem constants ----------------
#define NN 32
#define CC 64
#define TT 300
#define VV 25
#define SS 3
#define OCC 128
#define ICC 32
#define TVV 7500          // T*V
#define NTV 240000        // N*T*V
#define KE  9600          // IC*T
#define XE  15360000      // N*C*T*V

typedef __bf16 bf16x16 __attribute__((ext_vector_type(16)));
typedef __bf16 bf16x8  __attribute__((ext_vector_type(8)));
typedef float  f32x8   __attribute__((ext_vector_type(8)));

// ---------------- WMMA helpers (gfx1250, wave32) ----------------
__device__ __forceinline__ f32x8 wmma_bf16(bf16x16 a, bf16x16 b, f32x8 c) {
  return __builtin_amdgcn_wmma_f32_16x16x32_bf16(false, a, false, b, (short)0, c, false, false);
}

// A fragment (16x32 bf16): lane L = row m_base+(L&15); halves j: k = 16*(j>=8) + 8*(L>=16) + (j&7)
__device__ __forceinline__ bf16x16 load_a_frag(const __bf16* base, int ld, int m_base, int kc, int lane) {
  const int m  = m_base + (lane & 15);
  const int hi = (lane >> 4) & 1;
  const __bf16* p = base + (size_t)m * ld + kc + 8 * hi;
  bf16x8 lo = *(const bf16x8*)(p);
  bf16x8 hh = *(const bf16x8*)(p + 16);
  bf16x16 f;
#pragma unroll
  for (int j = 0; j < 8; ++j) { f[j] = lo[j]; f[j + 8] = hh[j]; }
  return f;
}

// same but rows not 16B-aligned (ld*2 not multiple of 16): scalar loads
__device__ __forceinline__ bf16x16 load_a_frag_u(const __bf16* base, int ld, int m_base, int kc, int lane) {
  const int m  = m_base + (lane & 15);
  const int hi = (lane >> 4) & 1;
  const __bf16* p = base + (size_t)m * ld + kc + 8 * hi;
  bf16x16 f;
#pragma unroll
  for (int j = 0; j < 8; ++j) { f[j] = p[j]; f[j + 8] = p[16 + j]; }
  return f;
}

// B fragment (32x16 bf16) from TRANSPOSED storage Bmem[col][k]: lane L = col n_base+(L&15),
// halves j: k = kc + 16*(j>=8) + 8*(L>=16) + (j&7)  (2x b128)
__device__ __forceinline__ bf16x16 load_b_frag(const __bf16* base, int ld, int n_base, int kc, int lane) {
  const int n = n_base + (lane & 15);
  const __bf16* p = base + (size_t)n * ld + kc + 8 * ((lane >> 4) & 1);
  bf16x8 lo = *(const bf16x8*)(p);
  bf16x8 hh = *(const bf16x8*)(p + 16);
  bf16x16 f;
#pragma unroll
  for (int j = 0; j < 8; ++j) { f[j] = lo[j]; f[j + 8] = hh[j]; }
  return f;
}

// ---------------- K0: cast x -> xb [n][c][t][v], xbT [n*7500+p][c] ----------------
__global__ void k_cast(const float* __restrict__ x, __bf16* __restrict__ xb, __bf16* __restrict__ xbT) {
  size_t tid = (size_t)blockIdx.x * 256 + threadIdx.x;   // exactly XE threads
  float v = x[tid];
  xb[tid] = (__bf16)v;
  int n = (int)(tid / 480000);
  int rem = (int)(tid % 480000);
  int c = rem / TVV, p = rem % TVV;
  xbT[((size_t)n * TVV + p) * 64 + c] = (__bf16)v;
}

// ---------------- K0b: weights -> bf16, bias sums ----------------
__global__ void k_prep(const float* caw, const float* cbw, const float* cdw, const float* rw,
                       const float* cdb, __bf16* We, __bf16* Wd, __bf16* Wr, float* bsum) {
  int tid = blockIdx.x * 256 + threadIdx.x;
  if (tid < 192 * 64) {                      // We[ch=192][c=64], ch = [S][a/b][IC]
    int ch = tid >> 6, c = tid & 63;
    int i = ch >> 6, ab = (ch >> 5) & 1, ic = ch & 31;
    const float* src = ab ? cbw : caw;
    We[tid] = (__bf16)src[(i * 32 + ic) * 64 + c];
  } else if (tid < 192 * 64 + 128 * 192) {   // Wd[oc][k=(i,c)=192]
    int u = tid - 192 * 64;
    int oc = u / 192, k = u % 192, i = k >> 6, c = k & 63;
    Wd[u] = (__bf16)cdw[((size_t)i * 128 + oc) * 64 + c];
  } else if (tid < 192 * 64 + 128 * 192 + 128 * 64) {
    int u = tid - (192 * 64 + 128 * 192);
    Wr[u] = (__bf16)rw[u];
  } else if (tid < 192 * 64 + 128 * 192 + 128 * 64 + 128) {
    int oc = tid - (192 * 64 + 128 * 192 + 128 * 64);
    bsum[oc] = cdb[oc] + cdb[128 + oc] + cdb[256 + oc];
  }
}

// ---------------- K1: e-GEMM  e[192ch][pos] = We @ xbT^T, scatter to eab[v][k] ----------------
__global__ void k_egemm(const __bf16* __restrict__ We, const __bf16* __restrict__ xbT,
                        const float* __restrict__ cab, const float* __restrict__ cbb,
                        __bf16* __restrict__ eab) {
  int lane = threadIdx.x & 31, wave = threadIdx.x >> 5;
  int mtile = blockIdx.x, n = blockIdx.z;
  int p0 = (blockIdx.y * 4 + wave) * 16;
  if (p0 >= TVV) return;                               // wave-uniform
  const __bf16* Bm = xbT + (size_t)n * TVV * 64;
  f32x8 acc = {};
#pragma unroll
  for (int kc = 0; kc < 64; kc += 32)
    acc = wmma_bf16(load_a_frag(We, 64, mtile * 16, kc, lane),
                    load_b_frag(Bm, 64, p0, kc, lane), acc);
  int hi = (lane >> 4) & 1, ln = lane & 15;
  int p = p0 + ln;
  if (p < TVV) {
    int t = p / 25, v = p % 25;
#pragma unroll
    for (int r = 0; r < 8; ++r) {
      int ch = mtile * 16 + 8 * hi + r;
      int i = ch >> 6, ab = (ch >> 5) & 1, ic = ch & 31;
      float bias = (ab ? cbb : cab)[i * 32 + ic];
      size_t idx = (((size_t)(i * 2 + ab) * NN + n) * VV + v) * KE + ic * TT + t;
      eab[idx] = (__bf16)(acc[r] + bias);
    }
  }
}

// ---------------- K2: P[n,v,w] = e1·e2ᵀ / V   (batched 32x32, K=9600, split-K over 8 waves) ----
__global__ void k_pgemm(const __bf16* __restrict__ eab, float* __restrict__ P) {
  __shared__ float red[8][32][32];
  int bi = blockIdx.x;                    // i*32+n
  int lane = threadIdx.x & 31, wave = threadIdx.x >> 5;
  const __bf16* A = eab + ((size_t)(2 * (bi / NN) + 0) * NN + (bi % NN)) * VV * KE;
  const __bf16* B = eab + ((size_t)(2 * (bi / NN) + 1) * NN + (bi % NN)) * VV * KE;
  f32x8 acc[2][2] = {{{}, {}}, {{}, {}}};
  for (int kc = wave * 32; kc < KE; kc += 8 * 32) {
    bf16x16 a0 = load_a_frag(A, KE, 0, kc, lane);
    bf16x16 a1 = load_a_frag(A, KE, 16, kc, lane);
    bf16x16 b0 = load_b_frag(B, KE, 0, kc, lane);
    bf16x16 b1 = load_b_frag(B, KE, 16, kc, lane);
    acc[0][0] = wmma_bf16(a0, b0, acc[0][0]);
    acc[0][1] = wmma_bf16(a0, b1, acc[0][1]);
    acc[1][0] = wmma_bf16(a1, b0, acc[1][0]);
    acc[1][1] = wmma_bf16(a1, b1, acc[1][1]);
  }
  int hi = (lane >> 4) & 1, ln = lane & 15;
#pragma unroll
  for (int mt = 0; mt < 2; ++mt)
#pragma unroll
    for (int nt = 0; nt < 2; ++nt)
#pragma unroll
      for (int r = 0; r < 8; ++r)
        red[wave][mt * 16 + 8 * hi + r][nt * 16 + ln] = acc[mt][nt][r];
  __syncthreads();
  for (int e = threadIdx.x; e < 1024; e += 256) {
    int m = e >> 5, w = e & 31;
    if (m < VV && w < VV) {
      float s = 0.f;
#pragma unroll
      for (int wv = 0; wv < 8; ++wv) s += red[wv][m][w];
      P[((size_t)bi * VV + m) * VV + w] = s * (1.0f / VV);
    }
  }
}

// ---------------- K3: column softmax + adj  ->  adaptT[w][v] bf16, zero-padded 32x32 ----------
__global__ void k_soft(const float* __restrict__ P, const float* __restrict__ adj,
                       const float* __restrict__ adjw, __bf16* __restrict__ adT) {
  int bi = blockIdx.x, i = bi / NN, w = threadIdx.x;
  __bf16* out = adT + (size_t)bi * 1024;
  if (w >= VV) {
    for (int v = 0; v < 32; ++v) out[w * 32 + v] = (__bf16)0.f;
    return;
  }
  const float* Pp = P + (size_t)bi * 625;
  float col[VV], mx = -1e30f;
  for (int v = 0; v < VV; ++v) { col[v] = Pp[v * 25 + w]; mx = fmaxf(mx, col[v]); }
  float s = 0.f;
  for (int v = 0; v < VV; ++v) { col[v] = __expf(col[v] - mx); s += col[v]; }
  float inv = 1.0f / s;
  for (int v = 0; v < VV; ++v)
    out[w * 32 + v] = (__bf16)(adj[i * 625 + v * 25 + w] + adjw[i * 625 + v * 25 + w] + col[v] * inv);
  for (int v = VV; v < 32; ++v) out[w * 32 + v] = (__bf16)0.f;
}

// ---------------- K4: z-GEMM  z[(c,t)][w] = x · adapt, scatter to zcat[pos][i*64+c] ------------
__global__ void k_zgemm(const __bf16* __restrict__ xb, const __bf16* __restrict__ adT,
                        __bf16* __restrict__ zcat) {
  int lane = threadIdx.x & 31, wave = threadIdx.x >> 5;
  int n = blockIdx.y, i = blockIdx.z;
  int m_base = (blockIdx.x * 4 + wave) * 16;              // row = c*300+t, 19200 rows
  const __bf16* A = xb + (size_t)n * 480000;              // ld = 25 (unaligned rows)
  const __bf16* Bm = adT + (size_t)(i * NN + n) * 1024;   // ld = 32, zero-padded
  bf16x16 a = load_a_frag_u(A, 25, m_base, 0, lane);      // K padded 25->32 (B zeros)
  f32x8 acc0 = {}, acc1 = {};
  acc0 = wmma_bf16(a, load_b_frag(Bm, 32, 0, 0, lane), acc0);
  acc1 = wmma_bf16(a, load_b_frag(Bm, 32, 16, 0, lane), acc1);
  int hi = (lane >> 4) & 1, ln = lane & 15;
#pragma unroll
  for (int r = 0; r < 8; ++r) {
    int mg = m_base + 8 * hi + r;
    int c = mg / TT, t = mg % TT;
    size_t rb = ((size_t)n * TVV + t * 25) * 192 + i * 64 + c;
    zcat[rb + (size_t)ln * 192] = (__bf16)acc0[r];        // w = ln < 25 always
    if (16 + ln < VV) zcat[rb + (size_t)(16 + ln) * 192] = (__bf16)acc1[r];
  }
}

// ---------------- K5: y-GEMM  y[oc][pos] = Wd[128][192] · zcat[pos][192]  -> d_out f32 ---------
__global__ void k_ygemm(const __bf16* __restrict__ Wd, const __bf16* __restrict__ zcat,
                        const float* __restrict__ bsum, float* __restrict__ y) {
  int lane = threadIdx.x & 31, wave = threadIdx.x >> 5;
  size_t p0 = (size_t)blockIdx.x * 16;                    // 240000/16 = 15000 exact
  const __bf16* Bm = zcat + p0 * 192;
  int oc0 = wave * 16;
  f32x8 acc = {};
#pragma unroll
  for (int kc = 0; kc < 192; kc += 32)
    acc = wmma_bf16(load_a_frag(Wd, 192, oc0, kc, lane),
                    load_b_frag(Bm, 192, 0, kc, lane), acc);
  int hi = (lane >> 4) & 1, ln = lane & 15;
  int pos = (int)p0 + ln, n = pos / TVV, p = pos % TVV;
#pragma unroll
  for (int r = 0; r < 8; ++r) {
    int oc = oc0 + 8 * hi + r;
    y[((size_t)n * 128 + oc) * TVV + p] = acc[r] + bsum[oc];
  }
}

// ---------------- K6: res-GEMM  res[oc][pos] = Wr[128][64] · xbT[pos][64]  -> rbuf bf16 --------
__global__ void k_rgemm(const __bf16* __restrict__ Wr, const __bf16* __restrict__ xbT,
                        const float* __restrict__ rb, __bf16* __restrict__ res) {
  int lane = threadIdx.x & 31, wave = threadIdx.x >> 5;
  size_t p0 = (size_t)blockIdx.x * 16;
  const __bf16* Bm = xbT + p0 * 64;
  int oc0 = wave * 16;
  f32x8 acc = {};
#pragma unroll
  for (int kc = 0; kc < 64; kc += 32)
    acc = wmma_bf16(load_a_frag(Wr, 64, oc0, kc, lane),
                    load_b_frag(Bm, 64, 0, kc, lane), acc);
  int hi = (lane >> 4) & 1, ln = lane & 15;
  int pos = (int)p0 + ln, n = pos / TVV, p = pos % TVV;
#pragma unroll
  for (int r = 0; r < 8; ++r) {
    int oc = oc0 + 8 * hi + r;
    res[((size_t)n * 128 + oc) * TVV + p] = (__bf16)(acc[r] + rb[oc]);
  }
}

// ---------------- K7: BN stats -> per-channel scale/shift for y and res ------------------------
__global__ void k_stats(const float* __restrict__ y, const __bf16* __restrict__ res,
                        const float* bn_g, const float* bn_b, const float* rs_g, const float* rs_b,
                        float* __restrict__ stats) {
  __shared__ float ss[256], sq[256];
  int which = blockIdx.x >> 7, ch = blockIdx.x & 127, tid = threadIdx.x;
  float s = 0.f, q = 0.f;
  for (int j = tid; j < NTV; j += 256) {
    int n = j / TVV, p = j % TVV;
    size_t idx = ((size_t)n * 128 + ch) * TVV + p;
    float v = which ? (float)res[idx] : y[idx];
    s += v; q += v * v;
  }
  ss[tid] = s; sq[tid] = q; __syncthreads();
  for (int o = 128; o > 0; o >>= 1) {
    if (tid < o) { ss[tid] += ss[tid + o]; sq[tid] += sq[tid + o]; }
    __syncthreads();
  }
  if (tid == 0) {
    float mean = ss[0] * (1.0f / NTV);
    float var = sq[0] * (1.0f / NTV) - mean * mean;
    float g = which ? rs_g[ch] : bn_g[ch];
    float b = which ? rs_b[ch] : bn_b[ch];
    float sc = g * rsqrtf(var + 1e-5f);
    stats[which * 256 + ch * 2]     = sc;
    stats[which * 256 + ch * 2 + 1] = b - mean * sc;
  }
}

// ---------------- K8: out = relu(bn(y) + bn(res)) ----------------------------------------------
__global__ void k_final(float* __restrict__ y, const __bf16* __restrict__ res,
                        const float* __restrict__ stats) {
  size_t t = (size_t)blockIdx.x * 256 + threadIdx.x;      // exactly 30,720,000
  int ch = (int)((t / TVV) % 128);
  float val = y[t] * stats[ch * 2] + stats[ch * 2 + 1]
            + (float)res[t] * stats[256 + ch * 2] + stats[256 + ch * 2 + 1];
  y[t] = fmaxf(val, 0.0f);
}

// ---------------- workspace layout ----------------
static constexpr size_t AL(size_t x) { return (x + 255) & ~(size_t)255; }
static constexpr size_t o_xb   = 0;                                        // XE bf16 + 64 pad
static constexpr size_t o_xbT  = AL(o_xb  + ((size_t)XE + 64) * 2);        // XE bf16 + 16*64 pad
static constexpr size_t o_eab  = AL(o_xbT + ((size_t)XE + 1024) * 2);      // S*2*N*V*KE + 8*KE pad
static constexpr size_t o_zc   = AL(o_eab + ((size_t)SS * 2 * NN * VV * KE + 8 * KE) * 2);
static constexpr size_t o_res  = AL(o_zc  + (size_t)NTV * 192 * 2);        // NTV*128 bf16
static constexpr size_t o_We   = AL(o_res + (size_t)NTV * 128 * 2);
static constexpr size_t o_Wd   = AL(o_We  + (size_t)192 * 64 * 2);
static constexpr size_t o_Wr   = AL(o_Wd  + (size_t)128 * 192 * 2);
static constexpr size_t o_bs   = AL(o_Wr  + (size_t)128 * 64 * 2);
static constexpr size_t o_P    = AL(o_bs  + 128 * 4);
static constexpr size_t o_adT  = AL(o_P   + (size_t)SS * NN * 625 * 4);
static constexpr size_t o_st   = AL(o_adT + (size_t)SS * NN * 1024 * 2);

extern "C" void kernel_launch(void* const* d_in, const int* in_sizes, int n_in,
                              void* d_out, int out_size, void* d_ws, size_t ws_size,
                              hipStream_t stream) {
  (void)in_sizes; (void)n_in; (void)out_size; (void)ws_size;
  const float* x    = (const float*)d_in[0];
  const float* adj  = (const float*)d_in[1];
  const float* adjw = (const float*)d_in[2];
  const float* caw  = (const float*)d_in[3];
  const float* cab  = (const float*)d_in[4];
  const float* cbw  = (const float*)d_in[5];
  const float* cbb  = (const float*)d_in[6];
  const float* cdw  = (const float*)d_in[7];
  const float* cdb  = (const float*)d_in[8];
  const float* rw   = (const float*)d_in[9];
  const float* rb   = (const float*)d_in[10];
  const float* rsg  = (const float*)d_in[11];
  const float* rsb  = (const float*)d_in[12];
  const float* bng  = (const float*)d_in[13];
  const float* bnb  = (const float*)d_in[14];
  float* out = (float*)d_out;

  char* ws = (char*)d_ws;
  __bf16* xb   = (__bf16*)(ws + o_xb);
  __bf16* xbT  = (__bf16*)(ws + o_xbT);
  __bf16* eab  = (__bf16*)(ws + o_eab);
  __bf16* zcat = (__bf16*)(ws + o_zc);
  __bf16* resb = (__bf16*)(ws + o_res);
  __bf16* We   = (__bf16*)(ws + o_We);
  __bf16* Wd   = (__bf16*)(ws + o_Wd);
  __bf16* Wr   = (__bf16*)(ws + o_Wr);
  float*  bsum = (float*)(ws + o_bs);
  float*  P    = (float*)(ws + o_P);
  __bf16* adT  = (__bf16*)(ws + o_adT);
  float*  st   = (float*)(ws + o_st);

  k_cast <<<XE / 256, 256, 0, stream>>>(x, xb, xbT);
  k_prep <<<177, 256, 0, stream>>>(caw, cbw, cdw, rw, cdb, We, Wd, Wr, bsum);
  k_egemm<<<dim3(12, 118, NN), 128, 0, stream>>>(We, xbT, cab, cbb, eab);
  k_pgemm<<<SS * NN, 256, 0, stream>>>(eab, P);
  k_soft <<<SS * NN, 32, 0, stream>>>(P, adj, adjw, adT);
  k_zgemm<<<dim3(300, NN, SS), 128, 0, stream>>>(xb, adT, zcat);
  k_ygemm<<<NTV / 16, 256, 0, stream>>>(Wd, zcat, bsum, out);
  k_rgemm<<<NTV / 16, 256, 0, stream>>>(Wr, xbT, rb, resb);
  k_stats<<<256, 256, 0, stream>>>(out, resb, bng, bnb, rsg, rsb, st);
  k_final<<<(size_t)NN * OCC * TVV / 256, 256, 0, stream>>>(out, resb, st);
}
